// TrajEncoder_43954695307840
// MI455X (gfx1250) — matrix operand order repaired and checked by reading, not compiled, not run on hardware
//
#include <hip/hip_runtime.h>
#include <math.h>

#define B_    16384
#define V_    256
#define L_    8
#define D_    3
#define HID   1024
#define OUT_  512
#define K_    5
#define INF_  24          // L*D
#define BK    (B_ * K_)   // 81920

typedef __attribute__((ext_vector_type(16))) __bf16  v16bf;
typedef __attribute__((ext_vector_type(8)))  float   v8f;
typedef __attribute__((ext_vector_type(4)))  unsigned int v4u;

struct W8 { unsigned int w[8]; };

static __device__ inline unsigned short f2bf(float f) {
  unsigned int u = __builtin_bit_cast(unsigned int, f);
  unsigned int r = u + 0x7FFFu + ((u >> 16) & 1u);   // RNE
  return (unsigned short)(r >> 16);
}
static __device__ inline unsigned int pack2(float a, float b) {
  return (unsigned int)f2bf(a) | ((unsigned int)f2bf(b) << 16);
}
static __device__ inline float bf2f(unsigned short h) {
  unsigned int u = ((unsigned int)h) << 16;
  return __builtin_bit_cast(float, u);
}
// Branch-free GELU (tanh form) on hardware transcendentals: no EXEC divergence.
static __device__ inline float gelu_fast(float x) {
  float u = 0.7978845608f * x * (1.0f + 0.044715f * x * x);
  float a = fabsf(u);
  float e = __builtin_amdgcn_exp2f(a * -2.885390082f);      // exp(-2a)
  float th = (1.0f - e) * __builtin_amdgcn_rcpf(1.0f + e);  // tanh(a)
  th = copysignf(th, u);
  return 0.5f * x * (1.0f + th);
}
// A-matrix K index used here (contiguous order) for VGPR word r: 2r + (lane<16?0:16)
static __device__ inline int a_kpat(int r, int lane) {
  return 2 * r + ((lane < 16) ? 0 : 16);
}

// ---------------------------------------------------------------------------
// Prep: norm_odo(vocab); pack w1 (24x1024, K padded to 32) and g⊙w2 (1024x512)
// into WMMA B-fragment layout: word index = ((kstep*32 + nt)*32 + lane)*8 + r
// B order used here (contiguous K): lane -> column nt*16+(lane&15); VGPR r -> K = 2r,2r+1
// (lanes 0-15) / 16+2r,16+2r+1 (lanes 16-31). Also fold beta@W2 + fc2 biases
// into one 512-wide output bias.
// ---------------------------------------------------------------------------
__global__ __launch_bounds__(256) void prep_kernel(
    const float* __restrict__ vocab,
    const float* __restrict__ w1v, const float* __restrict__ w1o,
    const float* __restrict__ w2v, const float* __restrict__ w2o,
    const float* __restrict__ v_ln_g, const float* __restrict__ v_ln_b,
    const float* __restrict__ o_ln_g, const float* __restrict__ o_ln_b,
    const float* __restrict__ v_fc2_b, const float* __restrict__ o_fc2_b,
    float* __restrict__ nvout,
    unsigned int* __restrict__ f1v, unsigned int* __restrict__ f1o,
    unsigned int* __restrict__ f2v, unsigned int* __restrict__ f2o,
    float* __restrict__ outbias) {
  int id = blockIdx.x * 256 + threadIdx.x;
  if (id < V_ * INF_) {                       // norm_odo
    int d = id % 3;
    float t = vocab[id];
    float r = (d == 0) ? 2.0f * (t + 1.57f) / 66.74f - 1.0f
            : (d == 1) ? 2.0f * (t + 19.68f) / 42.0f - 1.0f
                       : 2.0f * (t + 1.67f) / 3.53f - 1.0f;
    nvout[id] = r;
    return;
  }
  id -= V_ * INF_;
  if (id < 2 * 16384) {                       // w1 fragments (64 n-tiles, 1 k-step)
    const float* w1 = (id < 16384) ? w1v : w1o;
    unsigned int* f = (id < 16384) ? f1v : f1o;
    int q = id & 16383;
    int r = q & 7, lane = (q >> 3) & 31, nt = q >> 8;
    int col = nt * 16 + (lane & 15);
    int k = 2 * r + ((lane < 16) ? 0 : 16);
    float a = (k     < INF_) ? w1[k * HID + col]       : 0.0f;
    float b = (k + 1 < INF_) ? w1[(k + 1) * HID + col] : 0.0f;
    f[q] = pack2(a, b);
    return;
  }
  id -= 2 * 16384;
  if (id < 2 * 262144) {                      // (g ⊙ w2) fragments
    const float* w2 = (id < 262144) ? w2v : w2o;
    const float* lg = (id < 262144) ? v_ln_g : o_ln_g;
    unsigned int* f = (id < 262144) ? f2v : f2o;
    int q = id & 262143;
    int r = q & 7, lane = (q >> 3) & 31, nt = (q >> 8) & 31, ks = q >> 13;
    int col = nt * 16 + (lane & 15);
    int k = ks * 32 + 2 * r + ((lane < 16) ? 0 : 16);
    f[q] = pack2(lg[k] * w2[k * OUT_ + col], lg[k + 1] * w2[(k + 1) * OUT_ + col]);
    return;
  }
  id -= 2 * 262144;
  if (id < OUT_) {                            // folded output bias
    float s = v_fc2_b[id] + o_fc2_b[id];
    for (int k = 0; k < HID; ++k)
      s += v_ln_b[k] * w2v[k * OUT_ + id] + o_ln_b[k] * w2o[k * OUT_ + id];
    outbias[id] = s;
  }
}

// ---------------------------------------------------------------------------
// Top-K: one wave32 per trajectory b; vocab xy cached in LDS; 5 rounds of
// wave-wide argmin with tie-break on lowest index (matches top_k(-dist)).
// ---------------------------------------------------------------------------
__global__ __launch_bounds__(256) void topk_kernel(
    const float* __restrict__ traj, const float* __restrict__ vocab,
    int* __restrict__ idxout) {
  __shared__ float sv[V_ * L_ * 2];
  int t = threadIdx.x;
  for (int i = t; i < V_ * L_; i += 256) {
    sv[2 * i]     = vocab[3 * i];
    sv[2 * i + 1] = vocab[3 * i + 1];
  }
  __syncthreads();
  int wave = t >> 5, lane = t & 31;
  int b = blockIdx.x * 8 + wave;
  float tx[L_], ty[L_];
#pragma unroll
  for (int l = 0; l < L_; ++l) {
    tx[l] = traj[(b * L_ + l) * 3 + 0];
    ty[l] = traj[(b * L_ + l) * 3 + 1];
  }
  float dist[8];
  int   vid[8];
#pragma unroll
  for (int j = 0; j < 8; ++j) {
    int v = lane + 32 * j;
    float s = 0.0f;
#pragma unroll
    for (int l = 0; l < L_; ++l) {
      float dx = tx[l] - sv[2 * (v * L_ + l)];
      float dy = ty[l] - sv[2 * (v * L_ + l) + 1];
      s += sqrtf(dx * dx + dy * dy);
    }
    dist[j] = s;
    vid[j]  = v;
  }
  for (int k = 0; k < K_; ++k) {
    float bd = dist[0]; int bv = vid[0];
#pragma unroll
    for (int j = 1; j < 8; ++j)
      if (dist[j] < bd || (dist[j] == bd && vid[j] < bv)) { bd = dist[j]; bv = vid[j]; }
#pragma unroll
    for (int off = 16; off > 0; off >>= 1) {
      float od = __shfl_xor(bd, off, 32);
      int   ov = __shfl_xor(bv, off, 32);
      if (od < bd || (od == bd && ov < bv)) { bd = od; bv = ov; }
    }
#pragma unroll
    for (int j = 0; j < 8; ++j)
      if (vid[j] == bv) dist[j] = 3.0e38f;
    if (lane == 0) idxout[b * K_ + k] = bv;
  }
}

// ---------------------------------------------------------------------------
// Fused MLP: one block = 32 rows of BK, 256 threads = 8 waves
// (2 row-tiles x 4 n-splits). h tile is a single 64 KB XOR-swizzled bf16
// buffer: dword(row,k) = row*512 + ((k>>1) ^ row)  -> conflict-free padding-
// free fragment reads. Each (kstep,ntile) weight fragment feeds 2 WMMAs,
// halving L2 fragment traffic vs a 16-row tile. Both MLPs accumulate into the
// same WMMA C fragments (v_emb + o_emb).
// ---------------------------------------------------------------------------
__global__ __launch_bounds__(256) void mlp_kernel(
    const float* __restrict__ traj, const float* __restrict__ vocab,
    const float* __restrict__ nvoc, const int* __restrict__ idxb,
    const unsigned int* __restrict__ f1v, const unsigned int* __restrict__ f1o,
    const unsigned int* __restrict__ f2v, const unsigned int* __restrict__ f2o,
    const float* __restrict__ v_fc1_b, const float* __restrict__ o_fc1_b,
    const float* __restrict__ outbias, float* __restrict__ out) {
  __shared__ unsigned int shw[32 * 512];   // 64 KB swizzled bf16 h tile

  const int t    = threadIdx.x;
  const int wave = t >> 5, lane = t & 31;
  const int ml   = lane & 15;
  const int hb   = (lane < 16) ? 0 : 8;    // row-half offset for A/C fragments
  const int mt   = wave >> 2;              // row-tile 0/1
  const int ns   = wave & 3;               // n-split 0..3
  const int row0 = blockIdx.x * 32;

  // per-lane gather state for A1 fragments (row = mt*16 + ml)
  const int rowg = row0 + mt * 16 + ml;
  const int bidx = rowg / K_;
  const int vi   = idxb[rowg];

  v8f acc[8];
#pragma unroll
  for (int i = 0; i < 8; ++i) acc[i] = {};

  for (int p = 0; p < 2; ++p) {
    const float*        fc1b = p ? o_fc1_b : v_fc1_b;
    const unsigned int* f1   = p ? f1o     : f1v;
    const unsigned int* f2   = p ? f2o     : f2v;

    __syncthreads();                       // previous phase readers done

    // ---- A fragment for GEMM1: gather straight from global ----------------
    W8 aw;
#pragma unroll
    for (int r = 0; r < 8; ++r) {
      int k = a_kpat(r, lane);
      float x0 = 0.0f, x1 = 0.0f;
      if (k < INF_)
        x0 = p ? traj[bidx * INF_ + k] - vocab[vi * INF_ + k]
               : nvoc[vi * INF_ + k];
      if (k + 1 < INF_)
        x1 = p ? traj[bidx * INF_ + k + 1] - vocab[vi * INF_ + k + 1]
               : nvoc[vi * INF_ + k + 1];
      aw.w[r] = pack2(x0, x1);
    }
    v16bf afrag = __builtin_bit_cast(v16bf, aw);

    // ---- GEMM1 + bias + fast GELU -> swizzled h tile ----------------------
    for (int j = 0; j < 16; ++j) {
      int nt  = ns + 4 * j;                // 0..63 across the 4 n-splits
      int col = nt * 16 + ml;
      const v4u* bp = reinterpret_cast<const v4u*>(f1 + ((nt * 32 + lane) << 3));
      v4u b0 = bp[0], b1 = bp[1];
      W8 bw;
#pragma unroll
      for (int i = 0; i < 4; ++i) { bw.w[i] = b0[i]; bw.w[4 + i] = b1[i]; }
      v16bf bfrag = __builtin_bit_cast(v16bf, bw);
      v8f c = {};
      c = __builtin_amdgcn_wmma_f32_16x16x32_bf16(false, afrag, false, bfrag,
                                                  (short)0, c, false, false);
      float bias = fc1b[col];
      int dcol = col >> 1, half = col & 1;
      unsigned short* sh16 = reinterpret_cast<unsigned short*>(shw);
#pragma unroll
      for (int r = 0; r < 8; ++r) {
        int row = mt * 16 + r + hb;
        int dw  = row * 512 + (dcol ^ row);
        sh16[dw * 2 + half] = f2bf(gelu_fast(c[r] + bias));
      }
    }
    __syncthreads();

    // ---- per-row mean/var + in-place normalize (8 lanes per row, shuffles) -
    {
      int row = t >> 3, sub = t & 7;       // 8 consecutive lanes share a row
      unsigned int* hw = shw + row * 512;  // swizzle is a per-row bijection
      float s = 0.0f, s2 = 0.0f;
      for (int j = sub * 64; j < sub * 64 + 64; ++j) {
        unsigned int w = hw[j];
        float v0 = bf2f((unsigned short)w);
        float v1 = bf2f((unsigned short)(w >> 16));
        s += v0 + v1; s2 += v0 * v0 + v1 * v1;
      }
#pragma unroll
      for (int m = 1; m <= 4; m <<= 1) {
        s  += __shfl_xor(s,  m, 32);
        s2 += __shfl_xor(s2, m, 32);
      }
      float mu = s * (1.0f / HID);
      float rs = rsqrtf(s2 * (1.0f / HID) - mu * mu + 1e-5f);
      for (int j = sub * 64; j < sub * 64 + 64; ++j) {
        unsigned int w = hw[j];
        float v0 = (bf2f((unsigned short)w) - mu) * rs;
        float v1 = (bf2f((unsigned short)(w >> 16)) - mu) * rs;
        hw[j] = pack2(v0, v1);
      }
    }
    __syncthreads();

    // ---- GEMM2: A frags are pure ds_load_b32; preload B frags per k-step ---
    for (int kstep = 0; kstep < 32; ++kstep) {
      if (kstep + 1 < 32)
        __builtin_prefetch(f2 + (kstep + 1) * 8192 + ns * 2048, 0, 1);
      W8 a2;
#pragma unroll
      for (int r = 0; r < 8; ++r) {
        int k   = kstep * 32 + a_kpat(r, lane);   // always even
        int row = mt * 16 + ml;
        a2.w[r] = shw[row * 512 + ((k >> 1) ^ row)];
      }
      v16bf a2f = __builtin_bit_cast(v16bf, a2);
      v4u bl[16];
#pragma unroll
      for (int i = 0; i < 8; ++i) {
        const v4u* bp = reinterpret_cast<const v4u*>(
            f2 + (((kstep * 32 + ns * 8 + i) * 32 + lane) << 3));
        bl[2 * i]     = bp[0];
        bl[2 * i + 1] = bp[1];
      }
#pragma unroll
      for (int i = 0; i < 8; ++i) {
        W8 bw;
#pragma unroll
        for (int q = 0; q < 4; ++q) {
          bw.w[q]     = bl[2 * i][q];
          bw.w[4 + q] = bl[2 * i + 1][q];
        }
        v16bf bfrag = __builtin_bit_cast(v16bf, bw);
        acc[i] = __builtin_amdgcn_wmma_f32_16x16x32_bf16(
            false, a2f, false, bfrag, (short)0, acc[i], false, false);
      }
    }
  }

  // ---- epilogue: folded bias (fc2_v + fc2_o + beta@W2) and store ----------
#pragma unroll
  for (int i = 0; i < 8; ++i) {
    int col = (ns * 8 + i) * 16 + ml;
    float bias = outbias[col];
#pragma unroll
    for (int r = 0; r < 8; ++r)
      out[(size_t)(row0 + mt * 16 + r + hb) * OUT_ + col] = acc[i][r] + bias;
  }
}

// ---------------------------------------------------------------------------
extern "C" void kernel_launch(void* const* d_in, const int* in_sizes, int n_in,
                              void* d_out, int out_size, void* d_ws, size_t ws_size,
                              hipStream_t stream) {
  const float* traj    = (const float*)d_in[0];
  const float* vocab   = (const float*)d_in[1];
  const float* v_fc1_w = (const float*)d_in[2];
  const float* v_fc1_b = (const float*)d_in[3];
  const float* v_ln_g  = (const float*)d_in[4];
  const float* v_ln_b  = (const float*)d_in[5];
  const float* v_fc2_w = (const float*)d_in[6];
  const float* v_fc2_b = (const float*)d_in[7];
  const float* o_fc1_w = (const float*)d_in[8];
  const float* o_fc1_b = (const float*)d_in[9];
  const float* o_ln_g  = (const float*)d_in[10];
  const float* o_ln_b  = (const float*)d_in[11];
  const float* o_fc2_w = (const float*)d_in[12];
  const float* o_fc2_b = (const float*)d_in[13];
  float* out = (float*)d_out;

  char* ws = (char*)d_ws;
  // ws layout (256B aligned): idx | f1v | f1o | f2v | f2o | normvocab | outbias
  int*          idxb = (int*)ws;                                   // 327680 B
  unsigned int* f1v  = (unsigned int*)(ws + 327680);               // 65536 B
  unsigned int* f1o  = f1v + 16384;                                // 65536 B
  unsigned int* f2v  = (unsigned int*)(ws + 327680 + 131072);      // 1 MiB
  unsigned int* f2o  = f2v + 262144;                               // 1 MiB
  float*        nvoc = (float*)(ws + 327680 + 131072 + 2097152);   // 24576 B
  float*        obia = nvoc + V_ * INF_;                           // 2048 B

  // prep: 6144 + 32768 + 524288 + 512 = 563712 items -> 2202 blocks
  prep_kernel<<<2202, 256, 0, stream>>>(vocab, v_fc1_w, o_fc1_w, v_fc2_w, o_fc2_w,
                                        v_ln_g, v_ln_b, o_ln_g, o_ln_b,
                                        v_fc2_b, o_fc2_b,
                                        nvoc, f1v, f1o, f2v, f2o, obia);
  topk_kernel<<<B_ / 8, 256, 0, stream>>>(traj, vocab, idxb);
  mlp_kernel<<<BK / 32, 256, 0, stream>>>(traj, vocab, nvoc, idxb,
                                          f1v, f1o, f2v, f2o,
                                          v_fc1_b, o_fc1_b, obia, out);
}
